// EnergyAE_34170759807056
// MI455X (gfx1250) — matrix operand order repaired and checked by reading, not compiled, not run on hardware
//
#include <hip/hip_runtime.h>

#define NB 256   // batch
#define ND 3072  // data dim D
#define NH 2048  // hidden H
#define NZ 16    // latent n

typedef __attribute__((ext_vector_type(16))) __bf16 v16bf;
typedef __attribute__((ext_vector_type(8)))  __bf16 v8bf;
typedef __attribute__((ext_vector_type(8)))  float  v8f;

enum { EPI_NONE = 0, EPI_TANH = 1, EPI_RES = 2 };

__device__ __forceinline__ unsigned short f2bf(float f) {
    union { float f; unsigned u; } x; x.f = f;
    unsigned r = x.u + 0x7FFFu + ((x.u >> 16) & 1u);
    return (unsigned short)(r >> 16);
}
__device__ __forceinline__ float softplusf(float x) {
    return x > 20.f ? x : log1pf(expf(x));
}
__device__ __forceinline__ float sigmoidf(float x) {
    return 1.f / (1.f + expf(-x));
}

// ---------------------------------------------------------------------------
// bf16 WMMA GEMM:  C = epi( A(MxK) * B(KxN) )
//  - WM x WN waves / block; each wave owns NSUB independent 16x16 N-subtiles
//    (independent accumulators -> no WMMA->WMMA RAW chain, A-frag reuse)
//  - LDS tiles stored K-contiguous: A[row][k], B[col][k]; fragment loads are
//    2x ds_load_b128 per operand; staging is float4 global + ushort4 ds
//  - ASC/BSC: compile-time per-K scale of A/B (for P = V1 .* g etc.)
//  - BT: B stored row-major N x K (transposed access, k-contiguous in global)
//  - batched via blockIdx.z with element strides
// All launched shapes: M % (WM*16) == 0, N % (WN*16*NSUB) == 0, K % 32 == 0.
// ---------------------------------------------------------------------------
template<int WM, int WN, int NSUB, bool BT, bool ASC, bool BSC, int EPI>
__global__ __launch_bounds__(WM * WN * 32)
void gemm_wmma(const float* __restrict__ A, const float* __restrict__ Bm,
               float* __restrict__ C,
               int M, int N, int K, int lda, int ldb, int ldc,
               long sA, long sB, long sC,
               const float* __restrict__ ascale, long sAsc,
               const float* __restrict__ bscale, long sBsc,
               const float* __restrict__ bias,
               const float* __restrict__ X, long sX)
{
    constexpr int TM = WM * 16;
    constexpr int TN = WN * 16 * NSUB;
    constexpr int PA = 32 + 8;               // K-pitch (elements) for both tiles
    constexpr int NTHR = WM * WN * 32;
    __shared__ unsigned short As[TM * PA];   // [row][k]
    __shared__ unsigned short Bs[TN * PA];   // [col][k]

    const int bz = blockIdx.z;
    const float* Ab = A  + (long)bz * sA;
    const float* Bb = Bm + (long)bz * sB;
    float*       Cb = C  + (long)bz * sC;
    const float* asc = ASC ? ascale + (long)bz * sAsc : nullptr;
    const float* bsc = BSC ? bscale + (long)bz * sBsc : nullptr;

    const int tid  = threadIdx.x;
    const int wave = tid >> 5;
    const int lane = tid & 31;
    const int wm   = wave / WN;
    const int wn   = wave % WN;
    const int row0 = blockIdx.y * TM;
    const int col0 = blockIdx.x * TN;

    v8f acc[NSUB];
    #pragma unroll
    for (int j = 0; j < NSUB; ++j) acc[j] = v8f{};

    for (int k0 = 0; k0 < K; k0 += 32) {
        // ---- stage A tile (TM x 32), float4 loads -> ushort4 LDS stores ----
        constexpr int AV = (TM * 32) / (4 * NTHR);
        #pragma unroll
        for (int j = 0; j < AV; ++j) {
            int v  = tid + j * NTHR;
            int rw = v >> 3;                  // 8 float4 groups per 32-wide row
            int kc = (v & 7) << 2;
            float4 f = *(const float4*)(Ab + (long)(row0 + rw) * lda + k0 + kc);
            if constexpr (ASC) {
                float4 sv = *(const float4*)(asc + k0 + kc);
                f.x *= sv.x; f.y *= sv.y; f.z *= sv.z; f.w *= sv.w;
            }
            ushort4 o;
            o.x = f2bf(f.x); o.y = f2bf(f.y); o.z = f2bf(f.z); o.w = f2bf(f.w);
            *(ushort4*)(As + rw * PA + kc) = o;
        }
        // ---- stage B tile into [col][k] layout ----
        if constexpr (BT) {
            // B stored N x K: k-contiguous in global -> float4 along k
            constexpr int BV = (TN * 32) / (4 * NTHR);
            #pragma unroll
            for (int j = 0; j < BV; ++j) {
                int v  = tid + j * NTHR;
                int cw = v >> 3;
                int kc = (v & 7) << 2;
                float4 f = *(const float4*)(Bb + (long)(col0 + cw) * ldb + k0 + kc);
                if constexpr (BSC) {
                    float4 sv = *(const float4*)(bsc + k0 + kc);
                    f.x *= sv.x; f.y *= sv.y; f.z *= sv.z; f.w *= sv.w;
                }
                ushort4 o;
                o.x = f2bf(f.x); o.y = f2bf(f.y); o.z = f2bf(f.z); o.w = f2bf(f.w);
                *(ushort4*)(Bs + cw * PA + kc) = o;
            }
        } else {
            // B stored K x N: float4 along n, scalar transposed LDS stores
            constexpr int BV = (TN * 32) / (4 * NTHR);
            #pragma unroll
            for (int j = 0; j < BV; ++j) {
                int v  = tid + j * NTHR;
                int kk = v / (TN / 4);
                int nc = (v % (TN / 4)) << 2;
                float4 f = *(const float4*)(Bb + (long)(k0 + kk) * ldb + col0 + nc);
                if constexpr (BSC) {
                    float sv = bsc[k0 + kk];
                    f.x *= sv; f.y *= sv; f.z *= sv; f.w *= sv;
                }
                Bs[(nc + 0) * PA + kk] = f2bf(f.x);
                Bs[(nc + 1) * PA + kk] = f2bf(f.y);
                Bs[(nc + 2) * PA + kk] = f2bf(f.z);
                Bs[(nc + 3) * PA + kk] = f2bf(f.w);
            }
        }
        if (k0 + 32 < K && tid < TM)
            __builtin_prefetch(&Ab[(long)(row0 + tid) * lda + (k0 + 32)], 0, 1);
        __syncthreads();

        // ---- A fragment: two contiguous 8-elem K-runs -> 2x ds_load_b128 ----
        const int arow = wm * 16 + (lane & 15);
        const int ka   = (lane >> 4) << 3;     // 0 or 8
        v8bf alo = *(const v8bf*)(As + arow * PA + ka);
        v8bf ahi = *(const v8bf*)(As + arow * PA + 16 + ka);
        v16bf af = __builtin_shufflevector(alo, ahi,
                     0, 1, 2, 3, 4, 5, 6, 7, 8, 9, 10, 11, 12, 13, 14, 15);

        // ---- NSUB independent B fragments + WMMAs ----
        const int kb = (lane >> 4) << 3;       // 0 or 8
        #pragma unroll
        for (int j = 0; j < NSUB; ++j) {
            int bcol = wn * (16 * NSUB) + j * 16 + (lane & 15);
            v8bf blo = *(const v8bf*)(Bs + bcol * PA + kb);
            v8bf bhi = *(const v8bf*)(Bs + bcol * PA + 16 + kb);
            v16bf bfv = __builtin_shufflevector(blo, bhi,
                          0, 1, 2, 3, 4, 5, 6, 7, 8, 9, 10, 11, 12, 13, 14, 15);
            acc[j] = __builtin_amdgcn_wmma_f32_16x16x32_bf16(
                         false, af, false, bfv, (short)0, acc[j], false, false);
        }
        __syncthreads();
    }

    // ---- epilogue: C elem e -> M = e + (lane>=16)*8, N = lane%16 ----
    #pragma unroll
    for (int j = 0; j < NSUB; ++j) {
        const int cn = col0 + wn * (16 * NSUB) + j * 16 + (lane & 15);
        #pragma unroll
        for (int e = 0; e < 8; ++e) {
            int cm = row0 + wm * 16 + e + ((lane >> 4) ? 8 : 0);
            float v = acc[j][e];
            if constexpr (EPI == EPI_TANH) {
                v = tanhf(v + bias[cn]);
            } else if constexpr (EPI == EPI_RES) {
                v = X[(long)bz * sX + (long)cm * ldc + cn] - (v + bias[cn]);
            }
            Cb[(long)cm * ldc + cn] = v;
        }
    }
}

// ---- z_star = hx @ W2 + b2 : one block per sample ----
__global__ void enc2_kernel(const float* __restrict__ hx, const float* __restrict__ W2,
                            const float* __restrict__ b2, float* __restrict__ zst) {
    int b = blockIdx.x, tid = threadIdx.x;
    int k = tid & 15, g = tid >> 4;
    __shared__ float red[256];
    float s = 0.f;
    for (int m = g; m < NH; m += 16) s += hx[b * NH + m] * W2[m * NZ + k];
    red[tid] = s;
    __syncthreads();
    if (g == 0) {
        float t = 0.f;
        #pragma unroll
        for (int j = 0; j < 16; ++j) t += red[j * 16 + k];
        zst[b * NZ + k] = t + b2[k];
    }
}

// ---- h = tanh(z @ V1 + c1), g = 1 - h^2 : one thread per (b,m) ----
__global__ void dec_h_kernel(const float* __restrict__ z, const float* __restrict__ V1,
                             const float* __restrict__ c1,
                             float* __restrict__ h, float* __restrict__ g) {
    long i = (long)blockIdx.x * blockDim.x + threadIdx.x;
    int b = (int)(i / NH), m = (int)(i % NH);
    float a = c1[m];
    #pragma unroll
    for (int k = 0; k < NZ; ++k) a += z[b * NZ + k] * V1[k * NH + m];
    float t = tanhf(a);
    h[i] = t;
    if (g) g[i] = 1.f - t * t;
}

// ---- per-sample scalars: u,s,sigma(u),rho ; t = V1(g*Vsig) ; e = V1(g*w) ----
__global__ void scal1_kernel(const float* __restrict__ h, const float* __restrict__ g,
                             const float* __restrict__ w, const float* __restrict__ r,
                             const float* __restrict__ V1, const float* __restrict__ Vsig,
                             const float* __restrict__ csig,
                             float* __restrict__ tv, float* __restrict__ ev,
                             float* __restrict__ sc) {
    int b = blockIdx.x, tid = threadIdx.x;
    __shared__ float acc[64];
    if (tid < 64) acc[tid] = 0.f;
    __syncthreads();

    float u = 0.f, tk[NZ] = {}, ek[NZ] = {};
    for (int m = tid; m < NH; m += 256) {
        float hm = h[b * NH + m], gm = g[b * NH + m];
        float vs = Vsig[m], wm = w[b * NH + m];
        u += hm * vs;
        float gv = gm * vs, gw = gm * wm;
        #pragma unroll
        for (int k = 0; k < NZ; ++k) {
            float v1 = V1[k * NH + m];
            tk[k] += v1 * gv;
            ek[k] += v1 * gw;
        }
    }
    float rho = 0.f;
    for (int j = tid; j < ND; j += 256) { float rv = r[b * ND + j]; rho += rv * rv; }

    atomicAdd(&acc[0], u);
    atomicAdd(&acc[1], rho);
    #pragma unroll
    for (int k = 0; k < NZ; ++k) {
        atomicAdd(&acc[2 + k], tk[k]);
        atomicAdd(&acc[18 + k], ek[k]);
    }
    __syncthreads();
    if (tid == 0) {
        float uu = acc[0] + csig[0];
        float s  = softplusf(uu) + 1e-3f;
        sc[b * 8 + 0] = s;
        sc[b * 8 + 1] = sigmoidf(uu);
        sc[b * 8 + 2] = acc[1];
        sc[b * 8 + 3] = uu;
    }
    if (tid < NZ) {
        tv[b * NZ + tid] = acc[2 + tid];
        ev[b * NZ + tid] = acc[18 + tid];
    }
}

// ---- c[b][m] = (2/s^2) h g w - 2 sig (D/s - rho/s^3) h g Vsig ----
__global__ void cvec_kernel(const float* __restrict__ h, const float* __restrict__ g,
                            const float* __restrict__ w, const float* __restrict__ Vsig,
                            const float* __restrict__ sc, float* __restrict__ cv) {
    long i = (long)blockIdx.x * blockDim.x + threadIdx.x;
    int b = (int)(i / NH), m = (int)(i % NH);
    float s = sc[b * 8 + 0], sig = sc[b * 8 + 1], rho = sc[b * 8 + 2];
    float s2 = s * s, s3 = s2 * s;
    float hg = h[i] * g[i];
    cv[i] = (2.f / s2) * hg * w[i]
          - 2.f * sig * ((float)ND / s - rho / s3) * hg * Vsig[m];
}

// ---- assemble 16x16 Hessian, Jacobi eigvals, Cholesky, solve; 1 thr/sample --
__global__ void hess_kernel(const float* __restrict__ Gq, const float* __restrict__ Gc,
                            const float* __restrict__ tv, const float* __restrict__ ev,
                            const float* __restrict__ sc, const float* __restrict__ zst,
                            const float* __restrict__ eps,
                            float* __restrict__ zsm, float* __restrict__ ll) {
    int b = blockIdx.x * blockDim.x + threadIdx.x;
    if (b >= NB) return;
    float s = sc[b * 8 + 0], sig = sc[b * 8 + 1], rho = sc[b * 8 + 2];
    float s2 = s * s, s3 = s2 * s, s4 = s2 * s2;
    float Dv = (float)ND;
    float ct = (3.f * rho / s4 - Dv / s2) * sig * sig
             + (Dv / s - rho / s3) * sig * (1.f - sig);
    float ce = 2.f * sig / s3;

    float A[NZ][NZ], E[NZ][NZ], L[NZ][NZ];
    for (int k = 0; k < NZ; ++k)
        for (int l = 0; l < NZ; ++l) {
            float tk = tv[b * NZ + k], tl = tv[b * NZ + l];
            float ek_ = ev[b * NZ + k], el = ev[b * NZ + l];
            A[k][l] = Gq[b * 256 + k * NZ + l] / s2 + Gc[b * 256 + k * NZ + l]
                    + ce * (tl * ek_ + tk * el) + ct * tk * tl + (k == l ? 1.f : 0.f);
        }
    for (int k = 0; k < NZ; ++k)
        for (int l = 0; l < NZ; ++l) {
            float v = 0.5f * (A[k][l] + A[l][k]);
            A[k][l] = v; E[k][l] = v;
        }
    // cyclic Jacobi sweeps for eigenvalues
    for (int sweep = 0; sweep < 10; ++sweep)
        for (int p = 0; p < NZ; ++p)
            for (int q = p + 1; q < NZ; ++q) {
                float apq = E[p][q];
                if (fabsf(apq) < 1e-12f) continue;
                float theta = (E[q][q] - E[p][p]) / (2.f * apq);
                float tt = (theta >= 0.f ? 1.f : -1.f)
                         / (fabsf(theta) + sqrtf(theta * theta + 1.f));
                float cc = 1.f / sqrtf(tt * tt + 1.f);
                float ss = tt * cc;
                for (int i = 0; i < NZ; ++i) {
                    float eip = E[i][p], eiq = E[i][q];
                    E[i][p] = cc * eip - ss * eiq;
                    E[i][q] = ss * eip + cc * eiq;
                }
                for (int i = 0; i < NZ; ++i) {
                    float epi = E[p][i], eqi = E[q][i];
                    E[p][i] = cc * epi - ss * eqi;
                    E[q][i] = ss * epi + cc * eqi;
                }
            }
    float emin = E[0][0];
    for (int k = 1; k < NZ; ++k) emin = fminf(emin, E[k][k]);
    float delta = fmaxf(10.f - emin, 0.f);   // 1/MAX_LATENT_VARIANCE = 10
    float inv = 0.f, ld = 0.f;
    for (int k = 0; k < NZ; ++k) {
        float lam = E[k][k] + delta;
        inv += 1.f / lam;
        ld  += logf(lam);
    }
    // lower Cholesky of Prec = A + delta*I  (U = L^T, solve U y = eps)
    for (int k = 0; k < NZ; ++k) A[k][k] += delta;
    for (int j = 0; j < NZ; ++j) {
        float d = A[j][j];
        for (int p = 0; p < j; ++p) d -= L[j][p] * L[j][p];
        d = sqrtf(fmaxf(d, 1e-20f));
        L[j][j] = d;
        for (int i = j + 1; i < NZ; ++i) {
            float v = A[i][j];
            for (int p = 0; p < j; ++p) v -= L[i][p] * L[j][p];
            L[i][j] = v / d;
        }
    }
    float y[NZ];
    for (int i = NZ - 1; i >= 0; --i) {
        float v = eps[b * NZ + i];
        for (int j = i + 1; j < NZ; ++j) v -= L[j][i] * y[j];
        y[i] = v / L[i][i];
    }
    float zz = 0.f;
    for (int k = 0; k < NZ; ++k) {
        float zs = zst[b * NZ + k];
        zsm[b * NZ + k] = zs + y[k];
        zz += zs * zs;
    }
    ll[b] = 0.5f * zz + 0.5f * inv + 0.5f * ld;   // latent_energy + logdet
}

// ---- sigma at z_sample : one block per sample ----
__global__ void s2_kernel(const float* __restrict__ h2, const float* __restrict__ Vsig,
                          const float* __restrict__ csig, float* __restrict__ sc) {
    int b = blockIdx.x, tid = threadIdx.x;
    __shared__ float red;
    if (tid == 0) red = 0.f;
    __syncthreads();
    float u = 0.f;
    for (int m = tid; m < NH; m += 256) u += h2[b * NH + m] * Vsig[m];
    atomicAdd(&red, u);
    __syncthreads();
    if (tid == 0) sc[b * 8 + 4] = softplusf(red + csig[0]) + 1e-3f;
}

// ---- final: out = (recon + latent+logdet + D log s2) / D ----
__global__ void final_kernel(const float* __restrict__ rr, const float* __restrict__ sc,
                             const float* __restrict__ ll, float* __restrict__ out) {
    int b = blockIdx.x, tid = threadIdx.x;
    __shared__ float red;
    if (tid == 0) red = 0.f;
    __syncthreads();
    float a = 0.f;
    for (int j = tid; j < ND; j += 256) { float v = rr[b * ND + j]; a += v * v; }
    atomicAdd(&red, a);
    __syncthreads();
    if (tid == 0) {
        float s2 = sc[b * 8 + 4];
        float recon = red / (2.f * s2 * s2);
        out[b] = (recon + ll[b] + (float)ND * logf(s2)) / (float)ND;
    }
}

extern "C" void kernel_launch(void* const* d_in, const int* in_sizes, int n_in,
                              void* d_out, int out_size, void* d_ws, size_t ws_size,
                              hipStream_t stream) {
    (void)in_sizes; (void)n_in; (void)out_size; (void)ws_size;
    const float* x    = (const float*)d_in[0];
    const float* W1   = (const float*)d_in[1];
    const float* b1   = (const float*)d_in[2];
    const float* W2   = (const float*)d_in[3];
    const float* b2   = (const float*)d_in[4];
    const float* V1   = (const float*)d_in[5];
    const float* c1   = (const float*)d_in[6];
    const float* V2   = (const float*)d_in[7];
    const float* c2   = (const float*)d_in[8];
    const float* Vsig = (const float*)d_in[9];
    const float* csig = (const float*)d_in[10];
    const float* eps  = (const float*)d_in[11];
    float* out = (float*)d_out;

    float* p = (float*)d_ws;
    float* hx   = p; p += (long)NB * NH;
    float* hdec = p; p += (long)NB * NH;
    float* gdec = p; p += (long)NB * NH;
    float* w    = p; p += (long)NB * NH;
    float* cv   = p; p += (long)NB * NH;
    float* h2   = p; p += (long)NB * NH;
    float* Mm   = p; p += (long)NH * NH;
    float* PM   = p; p += (long)NB * NZ * NH;
    float* r    = p; p += (long)NB * ND;
    float* rr   = p; p += (long)NB * ND;
    float* zst  = p; p += NB * NZ;
    float* zsm  = p; p += NB * NZ;
    float* tv   = p; p += NB * NZ;
    float* ev   = p; p += NB * NZ;
    float* Gq   = p; p += NB * 256;
    float* Gc   = p; p += NB * 256;
    float* sc   = p; p += NB * 8;
    float* ll   = p; p += NB;

    // 1) hx = tanh(x @ W1 + b1)                    (256x2048, K=3072)
    gemm_wmma<4, 2, 2, false, false, false, EPI_TANH>
        <<<dim3(NH / 64, NB / 64, 1), 256, 0, stream>>>(
        x, W1, hx, NB, NH, ND, ND, NH, NH, 0, 0, 0,
        nullptr, 0, nullptr, 0, b1, nullptr, 0);
    // 2) z_star = hx @ W2 + b2
    enc2_kernel<<<NB, 256, 0, stream>>>(hx, W2, b2, zst);
    // 3) M = V2 @ V2^T                             (2048x2048, K=3072)
    gemm_wmma<4, 2, 2, true, false, false, EPI_NONE>
        <<<dim3(NH / 64, NH / 64, 1), 256, 0, stream>>>(
        V2, V2, Mm, NH, NH, ND, ND, ND, NH, 0, 0, 0,
        nullptr, 0, nullptr, 0, nullptr, nullptr, 0);
    // 4) h, g at z_star
    dec_h_kernel<<<(NB * NH) / 256, 256, 0, stream>>>(zst, V1, c1, hdec, gdec);
    // 5) r = x - (h @ V2 + c2)                     (256x3072, K=2048)
    gemm_wmma<4, 2, 2, false, false, false, EPI_RES>
        <<<dim3(ND / 64, NB / 64, 1), 256, 0, stream>>>(
        hdec, V2, r, NB, ND, NH, NH, ND, ND, 0, 0, 0,
        nullptr, 0, nullptr, 0, c2, x, 0);
    // 6) w = r @ V2^T                              (256x2048, K=3072)
    gemm_wmma<4, 2, 2, true, false, false, EPI_NONE>
        <<<dim3(NH / 64, NB / 64, 1), 256, 0, stream>>>(
        r, V2, w, NB, NH, ND, ND, ND, NH, 0, 0, 0,
        nullptr, 0, nullptr, 0, nullptr, nullptr, 0);
    // 7) per-sample scalars + t, e vectors
    scal1_kernel<<<NB, 256, 0, stream>>>(hdec, gdec, w, r, V1, Vsig, csig, tv, ev, sc);
    // 8) c vector for V1 diag(c) V1^T term
    cvec_kernel<<<(NB * NH) / 256, 256, 0, stream>>>(hdec, gdec, w, Vsig, sc, cv);
    // 9) PM_b = (V1 .* g_b) @ M   batched          (16x2048, K=2048, 256 batches)
    gemm_wmma<1, 4, 2, false, true, false, EPI_NONE>
        <<<dim3(NH / 128, 1, NB), 128, 0, stream>>>(
        V1, Mm, PM, NZ, NH, NH, NH, NH, NH, 0, 0, (long)NZ * NH,
        gdec, NH, nullptr, 0, nullptr, nullptr, 0);
    // 10) Gq_b = PM_b @ (V1 .* g_b)^T  batched     (16x16, K=2048)
    gemm_wmma<1, 1, 1, true, false, true, EPI_NONE>
        <<<dim3(1, 1, NB), 32, 0, stream>>>(
        PM, V1, Gq, NZ, NZ, NH, NH, NH, NZ, (long)NZ * NH, 0, 256,
        nullptr, 0, gdec, NH, nullptr, nullptr, 0);
    // 11) Gc_b = (V1 .* c_b) @ V1^T    batched     (16x16, K=2048)
    gemm_wmma<1, 1, 1, true, true, false, EPI_NONE>
        <<<dim3(1, 1, NB), 32, 0, stream>>>(
        V1, V1, Gc, NZ, NZ, NH, NH, NH, NZ, 0, 0, 256,
        cv, NH, nullptr, 0, nullptr, nullptr, 0);
    // 12) Hessian assembly + eig + Cholesky + solve -> z_sample, latent+logdet
    hess_kernel<<<1, 256, 0, stream>>>(Gq, Gc, tv, ev, sc, zst, eps, zsm, ll);
    // 13) h2 at z_sample
    dec_h_kernel<<<(NB * NH) / 256, 256, 0, stream>>>(zsm, V1, c1, h2, nullptr);
    // 14) s2 = softplus(h2 . Vsig + csig) + 1e-3
    s2_kernel<<<NB, 256, 0, stream>>>(h2, Vsig, csig, sc);
    // 15) rr = x - (h2 @ V2 + c2)
    gemm_wmma<4, 2, 2, false, false, false, EPI_RES>
        <<<dim3(ND / 64, NB / 64, 1), 256, 0, stream>>>(
        h2, V2, rr, NB, ND, NH, NH, ND, ND, 0, 0, 0,
        nullptr, 0, nullptr, 0, c2, x, 0);
    // 16) final loss
    final_kernel<<<NB, 256, 0, stream>>>(rr, sc, ll, out);
}
